// CRFLossTorch_23527830847809
// MI455X (gfx1250) — matrix operand order repaired and checked by reading, not compiled, not run on hardware
//
#include <hip/hip_runtime.h>
#include <hip/hip_bf16.h>
#include <stdint.h>

// ---------------------------------------------------------------------------
// CRF loss forward on MI455X (gfx1250).
//
//   * exp-domain recurrence: E[j,i]=exp(trans[j,i]-rowmax_j) precomputed;
//     per step: new_alpha[j] = m + rowmax_j + log(E[j,:].a') + feat_t[j]
//   * 16-workgroup cluster, 64 rows of E per WGP:
//       - staged global->LDS by the Tensor Data Mover (padded layout)
//       - K[256,1024) of each row held as fp32 in VGPRs (48/thread)
//       - K[0,256) held as f16 WMMA A-tiles (one 16x32 tile per wave)
//   * per step the matrix pipe (v_wmma_f32_16x16x32_f16, replicated-column B
//     loaded directly from an f16 copy of a') co-executes with the fp32 VALU
//     FMAs; alpha is exchanged via an L2 ping-pong buffer + cluster barrier.
//
// NOTE: dispatch should request a 16-workgroup cluster (cluster-launch
// extension); without it the cluster barrier is a NOP per the ISA.
// ---------------------------------------------------------------------------

#define TSTEPS   16384
#define NLAB     1024
#define START_TAG (NLAB - 2)
#define STOP_TAG  (NLAB - 1)

#define NWG      16          // workgroups in the cluster
#define ROWS     64          // labels (rows of E) per workgroup
#define BLOCK    1024        // threads per workgroup (32 waves, wave32)
#define SEGS     16          // threads cooperating per output row (VALU part)
#define KWMMA    256         // K range covered by the matrix pipe
#define KPTV     48          // fp32 K elements per thread ((1024-256)/16)
#define PSTR     1032        // padded LDS row stride, dwords (TDM pad +2dw/256dw)
#define APSTR    52          // padded a' chunk stride, dwords (48 + 4; 13x16B)

typedef unsigned int uint32x4 __attribute__((ext_vector_type(4)));
typedef int          int32x8  __attribute__((ext_vector_type(8)));
typedef int          int32x4  __attribute__((ext_vector_type(4)));
typedef _Float16     v16h     __attribute__((ext_vector_type(16)));
typedef float        v8f      __attribute__((ext_vector_type(8)));

// LDS dword index of element i within a padded E row (TDM pad layout)
__device__ __forceinline__ int eidx(int i) { return i + 2 * (i >> 8); }

__device__ __forceinline__ float waveRedMax(float v) {
  v = fmaxf(v, __shfl_xor(v, 16));
  v = fmaxf(v, __shfl_xor(v, 8));
  v = fmaxf(v, __shfl_xor(v, 4));
  v = fmaxf(v, __shfl_xor(v, 2));
  v = fmaxf(v, __shfl_xor(v, 1));
  return v;
}
__device__ __forceinline__ float waveRedSum(float v) {
  v += __shfl_xor(v, 16);
  v += __shfl_xor(v, 8);
  v += __shfl_xor(v, 4);
  v += __shfl_xor(v, 2);
  v += __shfl_xor(v, 1);
  return v;
}

__global__ __launch_bounds__(BLOCK, 1)
void crf_forward_kernel(const float* __restrict__ pred,   // [T, L]
                        const float* __restrict__ trans,  // [L, L]
                        float* __restrict__ ws)           // 2*L alpha + fs
{
  extern __shared__ float smem[];
  float* Es   = smem;                       // ROWS*PSTR : raw transitions slice
  float* aP   = Es + ROWS * PSTR;           // 16*APSTR  : padded fp32 a' (K>=256)
  float* aHf  = aP + SEGS * APSTR;          // 512 floats: f16 copy of a' (K all)
  float* rmax = aHf + 512;                  // ROWS
  float* red  = rmax + ROWS;                // 32
  float* part = red + 32;                   // ROWS*8 : wmma chunk partials

  const int tid  = threadIdx.x;
  const int wg   = blockIdx.x;
  const int jl   = tid >> 4;                // local row 0..63 (VALU ownership)
  const int seg  = tid & 15;                // VALU segment 0..15
  const int jg   = wg * ROWS + jl;
  const int wv   = tid >> 5;                // wave id 0..31
  const int lane = tid & 31;
  const bool laneHi = lane >= 16;

  // wmma task of this wave: row-tile rt (16 rows), K chunk kc (32 wide)
  const int rt = wv >> 3;
  const int kc = wv & 7;

  // a' writer index into the padded fp32 chunk layout (K in [256,1024))
  int aPidx = -1;
  if (tid >= KWMMA) {
    int p = tid - KWMMA;
    int c = p / KPTV;
    aPidx = c * APSTR + (p - c * KPTV);
  }

  // ---- stage this WGP's 64x1024 fp32 slice of `transitions` via TDM -------
#if __has_builtin(__builtin_amdgcn_tensor_load_to_lds)
  if (tid == 0) {
    uint64_t ga = (uint64_t)(uintptr_t)(trans + (size_t)wg * ROWS * NLAB);
    uint32_t la = (uint32_t)(uintptr_t)(void*)Es;     // low 32 bits = LDS offset
    uint32x4 g0 = { 1u, la, (uint32_t)ga,
                    (uint32_t)((ga >> 32) & 0x01FFFFFFu) | (2u << 30) };
    int32x8 g1;
    g1[0] = (2 << 16) | (1 << 20) | (7 << 22) | (1 << 25);  // 4B, pad 2dw/256dw
    g1[1] = (int)((unsigned)NLAB << 16);   // tensor_dim0 lo16
    g1[2] = (int)((unsigned)ROWS << 16);   // tensor_dim1 lo16
    g1[3] = (int)((unsigned)NLAB << 16);   // tile_dim0
    g1[4] = ROWS;                          // tile_dim1
    g1[5] = NLAB;                          // tensor_dim0_stride lo32
    g1[6] = 0;
    g1[7] = 0;
    int32x4 z4 = {0, 0, 0, 0};
    int32x8 z8 = {0, 0, 0, 0, 0, 0, 0, 0};
    __builtin_amdgcn_tensor_load_to_lds(g0, g1, z4, z4, z8, /*cpol=*/0);
    __builtin_amdgcn_s_wait_tensorcnt(0);
  }
#else
  for (int i = seg; i < NLAB; i += SEGS)
    Es[jl * PSTR + eidx(i)] = trans[(size_t)jg * NLAB + i];
#endif
  __syncthreads();

  // ---- per-row max of raw transitions --------------------------------------
  float rm = -3.4e38f;
  #pragma unroll 8
  for (int k = 0; k < NLAB / SEGS; ++k)
    rm = fmaxf(rm, Es[jl * PSTR + eidx(seg * (NLAB / SEGS) + k)]);
  rm = fmaxf(rm, __shfl_xor(rm, 8));
  rm = fmaxf(rm, __shfl_xor(rm, 4));
  rm = fmaxf(rm, __shfl_xor(rm, 2));
  rm = fmaxf(rm, __shfl_xor(rm, 1));
  if (seg == 0) rmax[jl] = rm;
  __syncthreads();
  const float rmj = rmax[jl];

  // fp32 E slice, K in [256,1024): 48 registers per thread
  float eReg[KPTV];
  #pragma unroll
  for (int k = 0; k < KPTV; ++k)
    eReg[k] = __expf(Es[jl * PSTR + eidx(KWMMA + seg * KPTV + k)] - rmj);

  // f16 WMMA A-tile, K in [0,256): one 16x32 tile per wave, 8 VGPRs/lane.
  // A layout (16-bit, 16x32): lanes 0-15 row M=lane hold K {0..7,16..23};
  // lanes 16-31 row M=lane-16 hold K {8..15,24..31}.
  const int rowA  = rt * 16 + (lane & 15);
  const int kbase = kc * 32;
  const float rmA = rmax[rowA];
  v16h aReg;
  #pragma unroll
  for (int h = 0; h < 16; ++h) {
    int K = kbase + ((h < 8) ? h : (8 + h)) + (laneHi ? 8 : 0);
    aReg[h] = (_Float16)__expf(Es[rowA * PSTR + eidx(K)] - rmA);
  }
  __syncthreads();

  // ---- sequential forward scan --------------------------------------------
  volatile float* abuf = (volatile float*)ws;   // ping-pong alpha [2][NLAB]
  const uint32_t* aH32 = (const uint32_t*)aHf;
  _Float16* aH16 = (_Float16*)aHf;
  const int bb = kc * 16 + (laneHi ? 4 : 0);    // B dwords for this wave/half

  for (int t = 0; t < TSTEPS; ++t) {
    const int wbase = (t & 1) * NLAB;
    const int rbase = ((t + 1) & 1) * NLAB;

    // m = max(alpha); a' = exp(alpha - m) stored fp32 (padded) + f16 copies
    float m;
    if (t == 0) {
      m = 0.0f;                                  // alpha0: 0 at START, NEG else
      float apv = (tid == START_TAG) ? 1.0f : 0.0f;
      aH16[tid] = (_Float16)apv;
      if (aPidx >= 0) aP[aPidx] = apv;
    } else {
      float av = abuf[rbase + tid];
      float wm = waveRedMax(av);
      if (lane == 0) red[wv] = wm;
      __syncthreads();
      m = red[0];
      #pragma unroll
      for (int w = 1; w < 32; ++w) m = fmaxf(m, red[w]);
      float apv = __expf(av - m);
      aH16[tid] = (_Float16)apv;
      if (aPidx >= 0) aP[aPidx] = apv;
    }
    __syncthreads();

    // --- matrix pipe: 16x32 f16 tile, B = a' chunk replicated across N -----
    uint32_t bw[8];
    #pragma unroll
    for (int v = 0; v < 8; ++v) bw[v] = aH32[bb + (v < 4 ? v : v + 4)];   // 2x ds_load_b128
    v16h bReg;
    __builtin_memcpy(&bReg, bw, sizeof(bReg));
    v8f cz = {0.f, 0.f, 0.f, 0.f, 0.f, 0.f, 0.f, 0.f};
    v8f d = __builtin_amdgcn_wmma_f32_16x16x32_f16(
        false, aReg, false, bReg, (short)0, cz, false, false);

    // --- VALU pipe (co-executes): fp32 dot over K in [256,1024) ------------
    float acc = 0.0f;
    #pragma unroll
    for (int k = 0; k < KPTV; ++k)
      acc = __builtin_fmaf(eReg[k], aP[seg * APSTR + k], acc);
    acc += __shfl_xor(acc, 8);
    acc += __shfl_xor(acc, 4);
    acc += __shfl_xor(acc, 2);
    acc += __shfl_xor(acc, 1);

    // scatter wmma chunk partials (all D columns identical; take N=0)
    if (lane == 0 || lane == 16) {
      int rowBase = rt * 16 + (laneHi ? 8 : 0);
      #pragma unroll
      for (int r = 0; r < 8; ++r) part[(rowBase + r) * 8 + kc] = d[r];
    }
    __syncthreads();

    if (seg == 0) {
      float wsum = 0.0f;
      #pragma unroll
      for (int q = 0; q < 8; ++q) wsum += part[jl * 8 + q];
      float feat = pred[(size_t)t * NLAB + jg];
      abuf[wbase + jg] = m + rmj + __logf(acc + wsum) + feat;
      if (t + 8 < TSTEPS)
        __builtin_prefetch(&pred[(size_t)(t + 8) * NLAB + jg], 0, 0);
    }

    __syncthreads();                       // protect aP/aH/red/part reuse
    __threadfence();                       // release alphas at device scope
#if __has_builtin(__builtin_amdgcn_s_cluster_barrier)
    __builtin_amdgcn_s_cluster_barrier();  // s_barrier_signal/wait -3
#endif
    __threadfence();                       // acquire next step's alphas
  }

  // ---- forward_score = logsumexp(alpha_T + transitions[STOP]) (wg 0) ------
  if (wg == 0) {
    float v = abuf[((TSTEPS - 1) & 1) * NLAB + tid] +
              trans[(size_t)STOP_TAG * NLAB + tid];
    float wm = waveRedMax(v);
    if (lane == 0) red[wv] = wm;
    __syncthreads();
    float M = red[0];
    #pragma unroll
    for (int w = 1; w < 32; ++w) M = fmaxf(M, red[w]);
    __syncthreads();
    float s = waveRedSum(__expf(v - M));
    if (lane == 0) red[wv] = s;
    __syncthreads();
    if (tid == 0) {
      float S = 0.0f;
      for (int w = 0; w < 32; ++w) S += red[w];
      ws[2 * NLAB] = M + __logf(S);       // forward score
    }
  }
}

// ---- gold path score + final subtraction ----------------------------------
__global__ __launch_bounds__(1024, 1)
void crf_gold_kernel(const float* __restrict__ pred,
                     const int* __restrict__ ref,
                     const float* __restrict__ trans,
                     const float* __restrict__ ws,
                     float* __restrict__ out)
{
  __shared__ float sred[32];
  const int tid = threadIdx.x;
  float g = 0.0f;
  for (int t = tid; t < TSTEPS; t += 1024) {
    int r = ref[t];
    int p = (t == 0) ? START_TAG : ref[t - 1];
    g += trans[(size_t)r * NLAB + p] + pred[(size_t)t * NLAB + r];
  }
  g = waveRedSum(g);
  if ((tid & 31) == 0) sred[tid >> 5] = g;
  __syncthreads();
  if (tid == 0) {
    float G = 0.0f;
    for (int w = 0; w < 32; ++w) G += sred[w];
    G += trans[(size_t)STOP_TAG * NLAB + ref[TSTEPS - 1]];
    out[0] = ws[2 * NLAB] - G;            // forward_score - gold_score
  }
}

extern "C" void kernel_launch(void* const* d_in, const int* in_sizes, int n_in,
                              void* d_out, int out_size, void* d_ws, size_t ws_size,
                              hipStream_t stream) {
  const float* pred  = (const float*)d_in[0];
  const int*   ref   = (const int*)d_in[1];
  const float* trans = (const float*)d_in[2];
  float* out = (float*)d_out;
  float* ws  = (float*)d_ws;

  const size_t shmem =
      (size_t)(ROWS * PSTR + SEGS * APSTR + 512 + ROWS + 32 + ROWS * 8) *
      sizeof(float);
  (void)hipFuncSetAttribute((const void*)crf_forward_kernel,
                            hipFuncAttributeMaxDynamicSharedMemorySize,
                            (int)shmem);

  crf_forward_kernel<<<dim3(NWG), dim3(BLOCK), shmem, stream>>>(pred, trans, ws);
  crf_gold_kernel<<<dim3(1), dim3(1024), 0, stream>>>(pred, ref, trans, ws, out);
}
